// BinaryDenseLayer_83288005804169
// MI455X (gfx1250) — matrix operand order repaired and checked by reading, not compiled, not run on hardware
//
#include <hip/hip_runtime.h>

typedef __attribute__((ext_vector_type(16))) __bf16 v16bf;
typedef __attribute__((ext_vector_type(8)))  float  v8f;
typedef __attribute__((ext_vector_type(4)))  float  v4f;

#define N_DIM 128           // N_IN == N_UNITS == 128
#define ROWS_PER_BLOCK 128  // 8 waves * 16 rows each
#define THREADS 256

__device__ __forceinline__ float sgnf(float w) {
  return (w > 0.f) ? 1.f : ((w < 0.f) ? -1.f : 0.f);
}

__device__ __forceinline__ unsigned int pack2bf(float a, float b) {
  __bf16 ba = (__bf16)a, bb = (__bf16)b;
  unsigned short ua = __builtin_bit_cast(unsigned short, ba);
  unsigned short ub = __builtin_bit_cast(unsigned short, bb);
  return (unsigned int)ua | ((unsigned int)ub << 16);
}

// out = x @ sign(W), computed as two bf16 WMMAs (hi/lo split of x) per K-chunk
// accumulating in f32. sign(W) is exact in bf16, so precision ~ f32.
// x and out are streamed once (no reuse) -> non-temporal loads/stores so the
// 2x134 MB stream doesn't rinse the 192 MB L2; only W (64 KB) is temporal.
__global__ __launch_bounds__(THREADS)
void binary_dense_wmma(const float* __restrict__ x,
                       const float* __restrict__ W,
                       float* __restrict__ out) {
  // B fragments in contiguous-K order (same as A): [kchunk][ntile][lane][dword]
  // VGPR j: lanes 0-15 hold K=2j,2j+1 ; lanes 16-31 hold K=16+2j,16+2j+1
  __shared__ __align__(16) unsigned int ldsB[4][8][32][8];  // 32 KB

  const int tid = threadIdx.x;

  // ---- Stage sign(W) (bf16) into LDS, pre-swizzled into B-fragment order ----
  for (int f = tid; f < 4 * 8 * 32; f += THREADS) {
    const int c    = f >> 8;          // K-chunk 0..3
    const int t    = (f >> 5) & 7;    // N-tile  0..7
    const int lane = f & 31;
    const int n    = (lane & 15) + t * 16;
    const int h    = lane >> 4;
#pragma unroll
    for (int j = 0; j < 8; ++j) {
      const int k0   = 32 * c + 16 * h + 2 * j;
      const float w0 = W[(size_t)k0 * N_DIM + n];
      const float w1 = W[(size_t)(k0 + 1) * N_DIM + n];
      ldsB[c][t][lane][j] = pack2bf(sgnf(w0), sgnf(w1));
    }
  }
  __syncthreads();

  const int wave = tid >> 5;
  const int lane = tid & 31;
  const int h    = lane >> 4;   // which K-half this lane covers
  const int m    = lane & 15;   // row within the 16-row tile / col within N-tile
  const long row0 = (long)blockIdx.x * ROWS_PER_BLOCK + (long)wave * 16;
  const long row  = row0 + m;

  const v8f vz = {0.f, 0.f, 0.f, 0.f, 0.f, 0.f, 0.f, 0.f};
  v8f acc[8];
#pragma unroll
  for (int t = 0; t < 8; ++t) acc[t] = vz;

#pragma unroll
  for (int c = 0; c < 4; ++c) {
    // A-fragment (16x32 bf16 layout): lane half h -> K = 32c+16h+{0..15};
    // 16 consecutive f32 = 4x b128 NT loads.
    const float* xr = x + row * N_DIM + 32 * c + 16 * h;
    const v4f p0 = __builtin_nontemporal_load((const v4f*)(xr + 0));
    const v4f p1 = __builtin_nontemporal_load((const v4f*)(xr + 4));
    const v4f p2 = __builtin_nontemporal_load((const v4f*)(xr + 8));
    const v4f p3 = __builtin_nontemporal_load((const v4f*)(xr + 12));

    float xs[16];
#pragma unroll
    for (int i = 0; i < 4; ++i) {
      xs[i]      = p0[i];
      xs[4 + i]  = p1[i];
      xs[8 + i]  = p2[i];
      xs[12 + i] = p3[i];
    }

    v16bf ahi, alo;
#pragma unroll
    for (int i = 0; i < 16; ++i) {
      const __bf16 hi = (__bf16)xs[i];
      ahi[i] = hi;
      alo[i] = (__bf16)(xs[i] - (float)hi);   // residual, exact split
    }

#pragma unroll
    for (int t = 0; t < 8; ++t) {
      const v16bf b = *(const v16bf*)(&ldsB[c][t][lane][0]);  // 2x ds_load_b128
      acc[t] = __builtin_amdgcn_wmma_f32_16x16x32_bf16(
          false, ahi, false, b, (short)0, acc[t], false, false);
      acc[t] = __builtin_amdgcn_wmma_f32_16x16x32_bf16(
          false, alo, false, b, (short)0, acc[t], false, false);
    }
  }

  // ---- Store: C layout -> lane l: col = (l&15)+16t, VGPR r: row = row0+8h+r
  float* orow = out + (row0 + 8 * h) * N_DIM + m;
#pragma unroll
  for (int t = 0; t < 8; ++t) {
#pragma unroll
    for (int r = 0; r < 8; ++r) {
      __builtin_nontemporal_store(acc[t][r], orow + (long)r * N_DIM + t * 16);
    }
  }
}

extern "C" void kernel_launch(void* const* d_in, const int* in_sizes, int n_in,
                              void* d_out, int out_size, void* d_ws, size_t ws_size,
                              hipStream_t stream) {
  const float* x = (const float*)d_in[0];
  const float* W = (const float*)d_in[1];
  float* out     = (float*)d_out;

  const int batch = in_sizes[0] / N_DIM;          // 262144
  dim3 grid(batch / ROWS_PER_BLOCK);              // 2048 blocks
  dim3 block(THREADS);                            // 8 waves (wave32)
  binary_dense_wmma<<<grid, block, 0, stream>>>(x, W, out);
}
